// TemplatePairStackBlock_17832704213162
// MI455X (gfx1250) — matrix operand order repaired and checked by reading, not compiled, not run on hardware
//
#include <hip/hip_runtime.h>
#include <hip/hip_bf16.h>
#include <cstdint>
#include <cstddef>

// ---------------------------------------------------------------------------
// Types / constants
// ---------------------------------------------------------------------------
typedef __attribute__((ext_vector_type(16))) __bf16 v16bf;
typedef __attribute__((ext_vector_type(8)))  float  v8f;

#define NRES   256
#define CT     64
#define MROWS  (NRES * NRES)     // 65536 rows per template
#define NHEADS 4
#define CATT   16
#define INF_F  1e9f

// ---------------------------------------------------------------------------
// WMMA helper: D = A(16x32 bf16) * B(32x16 bf16) + C(16x16 f32)
// ---------------------------------------------------------------------------
__device__ __forceinline__ v8f wmma_bf(v16bf a, v16bf b, v8f c) {
  return __builtin_amdgcn_wmma_f32_16x16x32_bf16(false, a, false, b,
                                                 (short)0, c, false, false);
}

// ---------------------------------------------------------------------------
// gfx1250 async global->LDS copy (ASYNCcnt path, ISA 08_async_tensor §4).
// LDS byte offset = low 32 bits of the generic pointer (flat->LDS aperture).
// ---------------------------------------------------------------------------
__device__ __forceinline__ void async_ld16(void* lds_dst, const void* gsrc) {
  unsigned lds = (unsigned)(uintptr_t)lds_dst;
  asm volatile("global_load_async_to_lds_b128 %0, %1, off"
               :: "v"(lds), "v"(gsrc) : "memory");
}
__device__ __forceinline__ void async_wait0() {
  asm volatile("s_wait_asynccnt 0x0" ::: "memory");
}

// ---------------------------------------------------------------------------
// Weight prep: f32 [K,N] row-major -> bf16 [Npad,K] (transposed, zero-padded)
// so that B fragments (lane = output column, contiguous K pairs) load as u32.
// ---------------------------------------------------------------------------
__global__ void prep_w_kernel(const float* __restrict__ src, __bf16* __restrict__ dst,
                              int K, int N, int Npad) {
  int idx = blockIdx.x * 256 + threadIdx.x;
  if (idx >= K * Npad) return;
  int n = idx / K;
  int k = idx - n * K;
  dst[idx] = (n < N) ? (__bf16)src[k * N + n] : (__bf16)0.0f;
}

// ---------------------------------------------------------------------------
// LayerNorm over last dim (64): one row per wave32, each lane owns 2 channels.
// Writes bf16 normalized activations.
// ---------------------------------------------------------------------------
__global__ void ln64_kernel(const float* __restrict__ x, const float* __restrict__ g,
                            const float* __restrict__ b, __bf16* __restrict__ out, int M) {
  int wid  = threadIdx.x >> 5;
  int lane = threadIdx.x & 31;
  int row  = blockIdx.x * 8 + wid;
  if (row >= M) return;
  const float2 v = ((const float2*)(x + (size_t)row * CT))[lane];
  float s  = v.x + v.y;
  float sq = v.x * v.x + v.y * v.y;
  #pragma unroll
  for (int d = 1; d < 32; d <<= 1) {
    s  += __shfl_xor(s,  d);
    sq += __shfl_xor(sq, d);
  }
  float mean = s  * (1.0f / 64.0f);
  float var  = sq * (1.0f / 64.0f) - mean * mean;
  float rstd = rsqrtf(var + 1e-5f);
  int c = lane * 2;
  float o0 = (v.x - mean) * rstd * g[c]     + b[c];
  float o1 = (v.y - mean) * rstd * g[c + 1] + b[c + 1];
  out[(size_t)row * CT + c]     = (__bf16)o0;
  out[(size_t)row * CT + c + 1] = (__bf16)o1;
}

// ---------------------------------------------------------------------------
// Generic WMMA GEMM:  C[M,N] = epi( A[M,K](bf16,row-major) * Wt[N,K](bf16) )
// Weight panel (<=16KB) is broadcast once per workgroup into LDS with the
// gfx1250 async copy path; 8 waves then read B fragments via ds_load.
// One 16x16 tile per wave, K-loop in steps of 32.
// ---------------------------------------------------------------------------
struct Epi {
  const float*  bias;      // [N] or null
  const __bf16* gate;      // [M,N] multiplier or null
  const float*  rowscale;  // [M] multiplier or null
  float*        outF;      // f32 output [M,N] or null
  __bf16*       outB;      // bf16 output [M,N] or null
  float         scale;
  int           act;       // 0 none, 1 sigmoid, 2 relu
  int           addF;      // 1: outF += v (residual)
};

__global__ void gemm_kernel(const __bf16* __restrict__ A, const __bf16* __restrict__ Wt,
                            int M, int N, int K, Epi e) {
  __shared__ __bf16 wsh[8192];                       // up to 128x64 bf16 panel
  int bytes = N * K * 2;
  for (int o = threadIdx.x * 16; o < bytes; o += 256 * 16)
    async_ld16((char*)wsh + o, (const char*)Wt + o);
  async_wait0();
  __syncthreads();

  int wid  = threadIdx.x >> 5;
  int lane = threadIdx.x & 31;
  int tile = blockIdx.x * 8 + wid;
  int ntiles = N >> 4;
  int mt = tile / ntiles;
  int nt = tile - mt * ntiles;
  if (mt * 16 >= M) return;
  int half = lane >> 4, l15 = lane & 15;

  v8f acc = {0.f, 0.f, 0.f, 0.f, 0.f, 0.f, 0.f, 0.f};
  const __bf16* arow = A + (size_t)(mt * 16 + l15) * K;
  __builtin_prefetch(arow, 0, 3);                    // global_prefetch_b8
  const __bf16* wrow = wsh + (size_t)(nt * 16 + l15) * K;
  for (int k0 = 0; k0 < K; k0 += 32) {
    v16bf a, b;
    #pragma unroll
    for (int v = 0; v < 8; ++v) {
      int ka = k0 + (v >> 2) * 16 + half * 8 + (v & 3) * 2;  // A frag layout
      a[2 * v]     = arow[ka];
      a[2 * v + 1] = arow[ka + 1];
      int kb = k0 + (v >> 2) * 16 + half * 8 + (v & 3) * 2;  // B frag layout
      b[2 * v]     = wrow[kb];
      b[2 * v + 1] = wrow[kb + 1];
    }
    acc = wmma_bf(a, b, acc);
  }
  #pragma unroll
  for (int r = 0; r < 8; ++r) {
    int row = mt * 16 + half * 8 + r;
    int col = nt * 16 + l15;
    float v = acc[r] * e.scale + (e.bias ? e.bias[col] : 0.f);
    if (e.act == 1)      v = 1.f / (1.f + __expf(-v));
    else if (e.act == 2) v = fmaxf(v, 0.f);
    if (e.gate)     v *= (float)e.gate[(size_t)row * N + col];
    if (e.rowscale) v *= e.rowscale[row];
    size_t o = (size_t)row * N + col;
    if (e.outF) { if (e.addF) e.outF[o] += v; else e.outF[o] = v; }
    if (e.outB) e.outB[o] = (__bf16)v;
  }
}

// ---------------------------------------------------------------------------
// Dual GEMM for triangle-mult projections:
//   out = rowscale[row] * sigmoid(A*W1 + b1) * (A*W2 + b2)   (bf16 output)
// Both weight panels staged into LDS via async copies (shared A fragment).
// ---------------------------------------------------------------------------
__global__ void gemm_dual_kernel(const __bf16* __restrict__ A,
                                 const __bf16* __restrict__ W1, const __bf16* __restrict__ W2,
                                 const float* __restrict__ b1, const float* __restrict__ b2,
                                 const float* __restrict__ rowscale,
                                 __bf16* __restrict__ outB, int M, int N, int K) {
  __shared__ __bf16 wsh[8192];                       // two 64x64 bf16 panels
  int nk = N * K;                                    // 4096
  for (int o = threadIdx.x * 16; o < nk * 2; o += 256 * 16)
    async_ld16((char*)wsh + o, (const char*)W1 + o);
  for (int o = threadIdx.x * 16; o < nk * 2; o += 256 * 16)
    async_ld16((char*)(wsh + nk) + o, (const char*)W2 + o);
  async_wait0();
  __syncthreads();

  int wid  = threadIdx.x >> 5;
  int lane = threadIdx.x & 31;
  int tile = blockIdx.x * 8 + wid;
  int ntiles = N >> 4;
  int mt = tile / ntiles;
  int nt = tile - mt * ntiles;
  if (mt * 16 >= M) return;
  int half = lane >> 4, l15 = lane & 15;

  v8f acc1 = {0.f, 0.f, 0.f, 0.f, 0.f, 0.f, 0.f, 0.f};
  v8f acc2 = {0.f, 0.f, 0.f, 0.f, 0.f, 0.f, 0.f, 0.f};
  const __bf16* arow = A + (size_t)(mt * 16 + l15) * K;
  __builtin_prefetch(arow, 0, 3);
  const __bf16* w1r = wsh      + (size_t)(nt * 16 + l15) * K;
  const __bf16* w2r = wsh + nk + (size_t)(nt * 16 + l15) * K;
  for (int k0 = 0; k0 < K; k0 += 32) {
    v16bf a, bb1, bb2;
    #pragma unroll
    for (int v = 0; v < 8; ++v) {
      int ka = k0 + (v >> 2) * 16 + half * 8 + (v & 3) * 2;
      a[2 * v]     = arow[ka];
      a[2 * v + 1] = arow[ka + 1];
      int kb = k0 + (v >> 2) * 16 + half * 8 + (v & 3) * 2;
      bb1[2 * v]     = w1r[kb];
      bb1[2 * v + 1] = w1r[kb + 1];
      bb2[2 * v]     = w2r[kb];
      bb2[2 * v + 1] = w2r[kb + 1];
    }
    acc1 = wmma_bf(a, bb1, acc1);
    acc2 = wmma_bf(a, bb2, acc2);
  }
  #pragma unroll
  for (int r = 0; r < 8; ++r) {
    int row = mt * 16 + half * 8 + r;
    int col = nt * 16 + l15;
    float gv = 1.f / (1.f + __expf(-(acc1[r] + b1[col])));
    float pv = acc2[r] + b2[col];
    float v  = gv * pv * rowscale[row];
    outB[(size_t)row * N + col] = (__bf16)v;
  }
}

// ---------------------------------------------------------------------------
// Triangle multiplication core:
//   o[i,j,c] = sum_k a[idx(i,k),c] * b[idx(j,k),c]
// One wave per (c, itile, jtile). Strides select outgoing / incoming variant.
// ---------------------------------------------------------------------------
__global__ void trimul_kernel(const __bf16* __restrict__ a, const __bf16* __restrict__ b,
                              float* __restrict__ o, int ai, int ak, int bj, int bk) {
  int wid  = threadIdx.x >> 5;
  int lane = threadIdx.x & 31;
  int tile = blockIdx.x * 8 + wid;          // 64 * 16 * 16 tiles
  int c   = tile >> 8;
  int rem = tile & 255;
  int it  = rem >> 4;
  int jt  = rem & 15;
  int half = lane >> 4, l15 = lane & 15;

  v8f acc = {0.f, 0.f, 0.f, 0.f, 0.f, 0.f, 0.f, 0.f};
  for (int k0 = 0; k0 < NRES; k0 += 32) {
    v16bf af, bf;
    #pragma unroll
    for (int v = 0; v < 8; ++v) {
      int ka = k0 + (v >> 2) * 16 + half * 8 + (v & 3) * 2;
      af[2 * v]     = a[(size_t)(it * 16 + l15) * ai + (size_t)ka * ak + c];
      af[2 * v + 1] = a[(size_t)(it * 16 + l15) * ai + (size_t)(ka + 1) * ak + c];
      int kb = k0 + (v >> 2) * 16 + half * 8 + (v & 3) * 2;
      bf[2 * v]     = b[(size_t)(jt * 16 + l15) * bj + (size_t)kb * bk + c];
      bf[2 * v + 1] = b[(size_t)(jt * 16 + l15) * bj + (size_t)(kb + 1) * bk + c];
    }
    acc = wmma_bf(af, bf, acc);
  }
  #pragma unroll
  for (int r = 0; r < 8; ++r) {
    int i = it * 16 + half * 8 + r;
    int j = jt * 16 + l15;
    o[((size_t)i * NRES + j) * CT + c] = acc[r];
  }
}

// ---------------------------------------------------------------------------
// Fused triangle attention (flash style). One wave per (i, h, jtile).
//   logits[j,k] = q[i,j]·k[i,k] (pre-scaled) + INF*(mask[i,k]-1) + tri[h,j,k]
//   o = softmax_k(logits) @ v ;  out = o * sigmoid-gate   (bf16, [M, H*D])
// QK^T: WMMA with d=16 zero-padded to K=32. P staged via LDS -> A fragment.
// ---------------------------------------------------------------------------
__global__ void attn_kernel(const __bf16* __restrict__ qb, const __bf16* __restrict__ kb,
                            const __bf16* __restrict__ vb, const __bf16* __restrict__ gb,
                            const __bf16* __restrict__ trib, const float* __restrict__ mask,
                            __bf16* __restrict__ ogb) {
  __shared__ __bf16 pls[8][16 * 32];
  int wid  = threadIdx.x >> 5;
  int lane = threadIdx.x & 31;
  int tile = blockIdx.x * 8 + wid;          // 256 * 4 * 16 tiles
  int jt = tile & 15;
  int h  = (tile >> 4) & 3;
  int i  = tile >> 6;
  int half = lane >> 4, l15 = lane & 15;

  // Hoisted Q fragment: rows j = jt*16 + l15, K-dim = d (16 real, 16 zero pad)
  v16bf aq;
  {
    const __bf16* qrow = qb + ((size_t)(i * NRES + jt * 16 + l15)) * CT + h * CATT;
    #pragma unroll
    for (int v = 0; v < 8; ++v) {
      int kk = (v >> 2) * 16 + half * 8 + (v & 3) * 2;
      if (kk < CATT) { aq[2 * v] = qrow[kk]; aq[2 * v + 1] = qrow[kk + 1]; }
      else           { aq[2 * v] = (__bf16)0.f; aq[2 * v + 1] = (__bf16)0.f; }
    }
  }

  v8f   oacc = {0.f, 0.f, 0.f, 0.f, 0.f, 0.f, 0.f, 0.f};
  float mrow[8], lrow[8];
  #pragma unroll
  for (int r = 0; r < 8; ++r) { mrow[r] = -1e30f; lrow[r] = 0.f; }

  for (int kt = 0; kt < NRES; kt += 32) {
    // --- logits for 32 key columns (two 16-wide WMMA tiles) ---
    v16bf bk0, bk1;
    #pragma unroll
    for (int v = 0; v < 8; ++v) {
      int dk = (v >> 2) * 16 + half * 8 + (v & 3) * 2;           // K-dim = d
      if (dk < CATT) {
        bk0[2 * v]     = kb[((size_t)(i * NRES + kt + l15)) * CT + h * CATT + dk];
        bk0[2 * v + 1] = kb[((size_t)(i * NRES + kt + l15)) * CT + h * CATT + dk + 1];
        bk1[2 * v]     = kb[((size_t)(i * NRES + kt + 16 + l15)) * CT + h * CATT + dk];
        bk1[2 * v + 1] = kb[((size_t)(i * NRES + kt + 16 + l15)) * CT + h * CATT + dk + 1];
      } else {
        bk0[2 * v] = (__bf16)0.f; bk0[2 * v + 1] = (__bf16)0.f;
        bk1[2 * v] = (__bf16)0.f; bk1[2 * v + 1] = (__bf16)0.f;
      }
    }
    v8f z = {0.f, 0.f, 0.f, 0.f, 0.f, 0.f, 0.f, 0.f};
    v8f s0 = wmma_bf(aq, bk0, z);
    v8f s1 = wmma_bf(aq, bk1, z);

    int   col0 = kt + l15, col1 = kt + 16 + l15;
    float mb0 = INF_F * (mask[i * NRES + col0] - 1.f);
    float mb1 = INF_F * (mask[i * NRES + col1] - 1.f);

    // --- online softmax over the 32 new columns ---
    #pragma unroll
    for (int r = 0; r < 8; ++r) {
      int j = jt * 16 + half * 8 + r;
      float t0 = s0[r] + mb0 + (float)trib[((size_t)j * NRES + col0) * 16 + h];
      float t1 = s1[r] + mb1 + (float)trib[((size_t)j * NRES + col1) * 16 + h];
      float mx = fmaxf(t0, t1);
      #pragma unroll
      for (int d = 1; d < 16; d <<= 1) mx = fmaxf(mx, __shfl_xor(mx, d));
      float mnew = fmaxf(mrow[r], mx);
      float sc = __expf(mrow[r] - mnew);
      float p0 = __expf(t0 - mnew);
      float p1 = __expf(t1 - mnew);
      float ps = p0 + p1;
      #pragma unroll
      for (int d = 1; d < 16; d <<= 1) ps += __shfl_xor(ps, d);
      lrow[r] = lrow[r] * sc + ps;
      mrow[r] = mnew;
      oacc[r] *= sc;
      int m = half * 8 + r;
      pls[wid][m * 32 + l15]      = (__bf16)p0;
      pls[wid][m * 32 + 16 + l15] = (__bf16)p1;
    }

    // --- O += P(16x32) @ V(32x16) --- (same-wave LDS RAW: DS ops in-order)
    v16bf ap, bv;
    #pragma unroll
    for (int v = 0; v < 8; ++v) {
      int kk = (v >> 2) * 16 + half * 8 + (v & 3) * 2;
      ap[2 * v]     = pls[wid][l15 * 32 + kk];
      ap[2 * v + 1] = pls[wid][l15 * 32 + kk + 1];
      int kv = (v >> 2) * 16 + half * 8 + (v & 3) * 2;
      bv[2 * v]     = vb[((size_t)(i * NRES + kt + kv))     * CT + h * CATT + l15];
      bv[2 * v + 1] = vb[((size_t)(i * NRES + kt + kv + 1)) * CT + h * CATT + l15];
    }
    oacc = wmma_bf(ap, bv, oacc);
  }

  // --- normalize, gate, store ---
  #pragma unroll
  for (int r = 0; r < 8; ++r) {
    int j = jt * 16 + half * 8 + r;
    size_t o = ((size_t)(i * NRES + j)) * CT + h * CATT + l15;
    float ov = oacc[r] / lrow[r];
    ogb[o] = (__bf16)(ov * (float)gb[o]);
  }
}

// ---------------------------------------------------------------------------
// Transposes: pair tensor [256,256,64] f32 (float4 granularity) and mask.
// ---------------------------------------------------------------------------
__global__ void transpose_z_kernel(const float* __restrict__ src, float* __restrict__ dst) {
  int idx = blockIdx.x * 256 + threadIdx.x;   // 256*256*16 float4 elements
  int c4 = idx & 15;
  int j  = (idx >> 4) & 255;
  int i  = idx >> 12;
  const float4* s = (const float4*)src;
  float4*       d = (float4*)dst;
  d[(size_t)(j * NRES + i) * 16 + c4] = s[(size_t)(i * NRES + j) * 16 + c4];
}

__global__ void transpose_mask_kernel(const float* __restrict__ src, float* __restrict__ dst) {
  int idx = blockIdx.x * 256 + threadIdx.x;
  if (idx >= MROWS) return;
  int r = idx >> 8, c = idx & 255;
  dst[(size_t)r * NRES + c] = src[(size_t)c * NRES + r];
}

// ---------------------------------------------------------------------------
// Host-side orchestration
// ---------------------------------------------------------------------------
struct AttW { const __bf16 *wq, *wk, *wv, *wg, *wtri, *wo;
              const float *ln_g, *ln_b, *bg, *bo; };
struct MulW { const __bf16 *wag, *wap, *wbg, *wbp, *wg, *wz;
              const float *ln_in_g, *ln_in_b, *bag, *bap, *bbg, *bbp,
                          *ln_out_g, *ln_out_b, *bgm, *bz; };
struct PtW  { const __bf16 *w1, *w2; const float *ln_g, *ln_b, *b1, *b2; };

struct Scratch {
  __bf16 *xnb, *b1, *b2, *b3, *b4, *trib, *hb;  // hb also holds ogb (first half)
  float  *fbuf, *maskT;
};

static Epi mkEpi() {
  Epi e; e.bias = nullptr; e.gate = nullptr; e.rowscale = nullptr;
  e.outF = nullptr; e.outB = nullptr; e.scale = 1.f; e.act = 0; e.addF = 0;
  return e;
}

static void launch_gemm(hipStream_t s, const __bf16* A, const __bf16* Wt,
                        int M, int N, int K, const Epi& e) {
  int tiles = (M / 16) * (N / 16);
  gemm_kernel<<<(tiles + 7) / 8, 256, 0, s>>>(A, Wt, M, N, K, e);
}

// z += tri_att(z) with given weights / mask (z: f32 [M,64], in-place residual)
static void run_att(hipStream_t s, float* z, const float* mask_t,
                    const AttW& w, const Scratch& sc) {
  ln64_kernel<<<MROWS / 8, 256, 0, s>>>(z, w.ln_g, w.ln_b, sc.xnb, MROWS);
  Epi e;
  e = mkEpi(); e.scale = 0.25f; e.outB = sc.b1;                 // q (pre-scaled)
  launch_gemm(s, sc.xnb, w.wq, MROWS, CT, CT, e);
  e = mkEpi(); e.outB = sc.b2;                                  // k
  launch_gemm(s, sc.xnb, w.wk, MROWS, CT, CT, e);
  e = mkEpi(); e.outB = sc.b3;                                  // v
  launch_gemm(s, sc.xnb, w.wv, MROWS, CT, CT, e);
  e = mkEpi(); e.act = 1; e.bias = w.bg; e.outB = sc.b4;        // gate
  launch_gemm(s, sc.xnb, w.wg, MROWS, CT, CT, e);
  e = mkEpi(); e.outB = sc.trib;                                // tri bias (N padded 16)
  launch_gemm(s, sc.xnb, w.wtri, MROWS, 16, CT, e);
  __bf16* ogb = sc.hb;
  attn_kernel<<<(NRES * NHEADS * 16) / 8, 256, 0, s>>>(sc.b1, sc.b2, sc.b3, sc.b4,
                                                       sc.trib, mask_t, ogb);
  e = mkEpi(); e.bias = w.bo; e.outF = z; e.addF = 1;           // out proj + residual
  launch_gemm(s, ogb, w.wo, MROWS, CT, CT, e);
}

// z += tri_mul(z) (outgoing or incoming)
static void run_mul(hipStream_t s, float* z, const float* mask_t,
                    const MulW& w, bool outgoing, const Scratch& sc) {
  ln64_kernel<<<MROWS / 8, 256, 0, s>>>(z, w.ln_in_g, w.ln_in_b, sc.xnb, MROWS);
  int tiles = (MROWS / 16) * (CT / 16);
  gemm_dual_kernel<<<(tiles + 7) / 8, 256, 0, s>>>(sc.xnb, w.wag, w.wap, w.bag, w.bap,
                                                   mask_t, sc.b1, MROWS, CT, CT);
  gemm_dual_kernel<<<(tiles + 7) / 8, 256, 0, s>>>(sc.xnb, w.wbg, w.wbp, w.bbg, w.bbp,
                                                   mask_t, sc.b2, MROWS, CT, CT);
  int ai = outgoing ? NRES * CT : CT;
  int ak = outgoing ? CT : NRES * CT;
  trimul_kernel<<<(CT * 16 * 16) / 8, 256, 0, s>>>(sc.b1, sc.b2, sc.fbuf, ai, ak, ai, ak);
  ln64_kernel<<<MROWS / 8, 256, 0, s>>>(sc.fbuf, w.ln_out_g, w.ln_out_b, sc.b3, MROWS);
  Epi e = mkEpi(); e.act = 1; e.bias = w.bgm; e.outB = sc.b4;   // output gate
  launch_gemm(s, sc.xnb, w.wg, MROWS, CT, CT, e);
  e = mkEpi(); e.bias = w.bz; e.gate = sc.b4; e.outF = z; e.addF = 1;
  launch_gemm(s, sc.b3, w.wz, MROWS, CT, CT, e);                // (o@wz+bz)*g + residual
}

// z += pair_transition(z)
static void run_pt(hipStream_t s, float* z, const float* mask_t,
                   const PtW& w, const Scratch& sc) {
  ln64_kernel<<<MROWS / 8, 256, 0, s>>>(z, w.ln_g, w.ln_b, sc.xnb, MROWS);
  Epi e = mkEpi(); e.act = 2; e.bias = w.b1; e.outB = sc.hb;    // relu(x@w1+b1), N=128
  launch_gemm(s, sc.xnb, w.w1, MROWS, 2 * CT, CT, e);
  e = mkEpi(); e.bias = w.b2; e.rowscale = mask_t; e.outF = z; e.addF = 1;
  launch_gemm(s, sc.hb, w.w2, MROWS, CT, 2 * CT, e);            // (h@w2+b2)*mask + res
}

// Param flat-index map (insertion-order depth-first flatten of setup_inputs)
enum { IN_Z = 0, IN_MASK = 1, TAS = 2, TAE = 12, TMO = 22, TMI = 38, PTI = 54 };

extern "C" void kernel_launch(void* const* d_in, const int* in_sizes, int n_in,
                              void* d_out, int out_size, void* d_ws, size_t ws_size,
                              hipStream_t stream) {
  (void)in_sizes; (void)n_in; (void)out_size; (void)ws_size;

  const float* z_in = (const float*)d_in[IN_Z];
  const float* mask = (const float*)d_in[IN_MASK];
  float* z = (float*)d_out;                          // [T, 256, 256, 64]
  hipMemcpyAsync(z, z_in, sizeof(float) * 2 * (size_t)MROWS * CT,
                 hipMemcpyDeviceToDevice, stream);

  // ---- scratch carve ----
  char* wsb = (char*)d_ws;
  size_t off = 0;
  auto carve = [&](size_t bytes) -> void* {
    void* p = wsb + off; off += (bytes + 255) & ~(size_t)255; return p;
  };
  auto prep = [&](int idx, int K, int N, int Npad) -> const __bf16* {
    __bf16* dst = (__bf16*)carve((size_t)K * Npad * 2);
    int tot = K * Npad;
    prep_w_kernel<<<(tot + 255) / 256, 256, 0, stream>>>((const float*)d_in[idx],
                                                         dst, K, N, Npad);
    return dst;
  };

  AttW aw[2]; // tas, tae
  int abase[2] = {TAS, TAE};
  for (int a = 0; a < 2; ++a) {
    int b = abase[a];
    aw[a].ln_g = (const float*)d_in[b + 0];
    aw[a].ln_b = (const float*)d_in[b + 1];
    aw[a].wq   = prep(b + 2, CT, CT, CT);
    aw[a].wk   = prep(b + 3, CT, CT, CT);
    aw[a].wv   = prep(b + 4, CT, CT, CT);
    aw[a].wtri = prep(b + 5, CT, NHEADS, 16);
    aw[a].wg   = prep(b + 6, CT, CT, CT);
    aw[a].bg   = (const float*)d_in[b + 7];
    aw[a].wo   = prep(b + 8, CT, CT, CT);
    aw[a].bo   = (const float*)d_in[b + 9];
  }
  MulW mw[2]; // tmo, tmi
  int mbase[2] = {TMO, TMI};
  for (int a = 0; a < 2; ++a) {
    int b = mbase[a];
    mw[a].ln_in_g  = (const float*)d_in[b + 0];
    mw[a].ln_in_b  = (const float*)d_in[b + 1];
    mw[a].wag      = prep(b + 2, CT, CT, CT);
    mw[a].bag      = (const float*)d_in[b + 3];
    mw[a].wap      = prep(b + 4, CT, CT, CT);
    mw[a].bap      = (const float*)d_in[b + 5];
    mw[a].wbg      = prep(b + 6, CT, CT, CT);
    mw[a].bbg      = (const float*)d_in[b + 7];
    mw[a].wbp      = prep(b + 8, CT, CT, CT);
    mw[a].bbp      = (const float*)d_in[b + 9];
    mw[a].ln_out_g = (const float*)d_in[b + 10];
    mw[a].ln_out_b = (const float*)d_in[b + 11];
    mw[a].wg       = prep(b + 12, CT, CT, CT);
    mw[a].bgm      = (const float*)d_in[b + 13];
    mw[a].wz       = prep(b + 14, CT, CT, CT);
    mw[a].bz       = (const float*)d_in[b + 15];
  }
  PtW pw;
  pw.ln_g = (const float*)d_in[PTI + 0];
  pw.ln_b = (const float*)d_in[PTI + 1];
  pw.w1   = prep(PTI + 2, CT, 2 * CT, 2 * CT);
  pw.b1   = (const float*)d_in[PTI + 3];
  pw.w2   = prep(PTI + 4, 2 * CT, CT, CT);
  pw.b2   = (const float*)d_in[PTI + 5];

  Scratch sc;
  sc.maskT = (float*)carve((size_t)MROWS * 4);
  sc.xnb   = (__bf16*)carve((size_t)MROWS * CT * 2);
  sc.b1    = (__bf16*)carve((size_t)MROWS * CT * 2);
  sc.b2    = (__bf16*)carve((size_t)MROWS * CT * 2);
  sc.b3    = (__bf16*)carve((size_t)MROWS * CT * 2);
  sc.b4    = (__bf16*)carve((size_t)MROWS * CT * 2);
  sc.trib  = (__bf16*)carve((size_t)MROWS * 16 * 2);
  sc.hb    = (__bf16*)carve((size_t)MROWS * 2 * CT * 2);
  sc.fbuf  = (float*)carve((size_t)MROWS * CT * 4);

  // ---- five sublayers per template, sequential residual updates ----
  for (int t = 0; t < 2; ++t) {
    float*       zt     = z + (size_t)t * MROWS * CT;
    const float* mask_t = mask + (size_t)t * MROWS;

    // 1) triangle attention (starting node)
    run_att(stream, zt, mask_t, aw[0], sc);

    // 2) triangle attention (ending node): operate on transposed pair tensor
    transpose_z_kernel<<<(MROWS * 16) / 256, 256, 0, stream>>>(zt, sc.fbuf);
    transpose_mask_kernel<<<MROWS / 256, 256, 0, stream>>>(mask_t, sc.maskT);
    run_att(stream, sc.fbuf, sc.maskT, aw[1], sc);
    transpose_z_kernel<<<(MROWS * 16) / 256, 256, 0, stream>>>(sc.fbuf, zt);

    // 3) triangle multiplication (outgoing)
    run_mul(stream, zt, mask_t, mw[0], true, sc);
    // 4) triangle multiplication (incoming)
    run_mul(stream, zt, mask_t, mw[1], false, sc);
    // 5) pair transition
    run_pt(stream, zt, mask_t, pw, sc);
  }
}
